// RelGraphConvLayer_2405181685904
// MI455X (gfx1250) — matrix operand order repaired and checked by reading, not compiled, not run on hardware
//
#include <hip/hip_runtime.h>

typedef __attribute__((ext_vector_type(16))) __bf16 v16bf;
typedef __attribute__((ext_vector_type(8)))  float  v8f;

#define DMODEL 128

__device__ __forceinline__ unsigned short f2bf(float f) {
  unsigned int u = __float_as_uint(f);
  u += 0x7FFFu + ((u >> 16) & 1u);          // round-to-nearest-even
  return (unsigned short)(u >> 16);
}

__global__ void k_zero_f32(float* __restrict__ p, long n) {
  long i = (long)blockIdx.x * blockDim.x + threadIdx.x;
  if (i < n) p[i] = 0.0f;
}

__global__ void k_cvt_bf16(const float* __restrict__ s, unsigned short* __restrict__ d, long n) {
  long i = (long)blockIdx.x * blockDim.x + threadIdx.x;
  if (i < n) d[i] = f2bf(s[i]);
}

// Transpose weights to [n][k] bf16 so B fragments are contiguous 32B runs along K.
// Slot R holds the self-loop weight Wl.
__global__ void k_cvt_w_t(const float* __restrict__ W, const float* __restrict__ Wl,
                          unsigned short* __restrict__ wbt, int R) {
  int t = blockIdx.x * blockDim.x + threadIdx.x;
  int total = (R + 1) * DMODEL * DMODEL;
  if (t >= total) return;
  int rel = t / (DMODEL * DMODEL);
  int idx = t % (DMODEL * DMODEL);
  int k = idx / DMODEL, n = idx % DMODEL;
  float v = (rel < R) ? W[(size_t)rel * DMODEL * DMODEL + idx] : Wl[idx];
  wbt[(size_t)rel * DMODEL * DMODEL + (size_t)n * DMODEL + k] = f2bf(v);
}

__global__ void k_degree(const int* __restrict__ dst, float* __restrict__ deg,
                         int RE, int E, int N) {
  int e = blockIdx.x * blockDim.x + threadIdx.x;
  if (e >= RE) return;
  int r = e / E;
  atomicAdd(&deg[(size_t)r * N + dst[e]], 1.0f);
}

__global__ void k_invdeg(float* __restrict__ deg, long n) {
  long i = (long)blockIdx.x * blockDim.x + threadIdx.x;
  if (i < n) deg[i] = 1.0f / fmaxf(deg[i], 1.0f);
}

// ---- bf16 WMMA GEMM, 8 waves/block, one 16x128 output slab per wave.
// B (128x128 bf16, [n][k] pre-transposed) is staged once per block into LDS;
// A slab (16x128 per wave) is preloaded into registers in one clause.
// B fragments are software-pipelined (bcur/bnext ping-pong) so a fragment's
// two ds_load_b128 are in flight while the previous WMMA executes.
// Fragment layouts used here (A and B share the same contiguous K order):
//   A (16-bit 16x32): lane m=lane&15, half=lane>>4; 8 VGPRs = 16 contiguous K starting at kk+16*half
//   B (16-bit 32x16): lane n=lane&15; 8 VGPRs = 16 contiguous K starting at kk+16*half
//   C (f32 16x16):    VGPR v -> row v + 8*half, col lane&15
template <bool FUSE_FINAL>
__global__ __launch_bounds__(256)
void k_gemm_bf16(const unsigned short* __restrict__ A, const unsigned short* __restrict__ Bt,
                 float* __restrict__ Y, const float* __restrict__ bias, int Nrows) {
  __shared__ unsigned short sB[DMODEL * DMODEL];   // 32 KB of 320 KB/WGP
  {
    const uint4* g = (const uint4*)Bt;             // 16384 bf16 = 2048 uint4
    uint4* s = (uint4*)sB;
    #pragma unroll
    for (int i = 0; i < 8; i++)
      s[threadIdx.x + i * 256] = g[threadIdx.x + i * 256];
  }
  __syncthreads();

  int waveInBlk = threadIdx.x >> 5;
  int lane = threadIdx.x & 31;
  int m0 = (blockIdx.x * 8 + waveInBlk) * 16;
  if (m0 >= Nrows) return;                          // uniform per-wave exit
  int half = lane >> 4, l16 = lane & 15;

  // Preload the full A slab for this wave: 4 k-fragments (32 VGPRs), one clause.
  const uint4* arow = (const uint4*)(A + (size_t)(m0 + l16) * DMODEL);  // uint4 = 8 bf16
  union { v16bf v; uint4 u[2]; } af[4];
  #pragma unroll
  for (int kk4 = 0; kk4 < 4; kk4++) {
    af[kk4].u[0] = arow[kk4 * 4 + 2 * half];        // K = 32*kk4 + 16h .. +7
    af[kk4].u[1] = arow[kk4 * 4 + 2 * half + 1];    // K = 32*kk4 + 16h + 8 .. +15
  }

  v8f c[8] = {};
  #pragma unroll
  for (int kk4 = 0; kk4 < 4; kk4++) {
    const unsigned short* bbase = sB + l16 * DMODEL + kk4 * 32 + 16 * half;
    v16bf bcur = *(const v16bf*)bbase;
    #pragma unroll
    for (int nt = 0; nt < 8; nt++) {
      v16bf bnext = bcur;
      if (nt < 7) bnext = *(const v16bf*)(bbase + (nt + 1) * 16 * DMODEL);
      c[nt] = __builtin_amdgcn_wmma_f32_16x16x32_bf16(false, af[kk4].v, false, bcur,
                                                      (short)0, c[nt], false, false);
      bcur = bnext;
    }
  }

  if (FUSE_FINAL) {
    #pragma unroll
    for (int nt = 0; nt < 8; nt++) {
      float bcol = bias[nt * 16 + l16];
      #pragma unroll
      for (int v = 0; v < 8; v++) {
        size_t idx = (size_t)(m0 + v + 8 * half) * DMODEL + nt * 16 + l16;
        Y[idx] = fmaxf(c[nt][v] + Y[idx] + bcol, 0.0f);
      }
    }
  } else {
    #pragma unroll
    for (int nt = 0; nt < 8; nt++)
      #pragma unroll
      for (int v = 0; v < 8; v++)
        Y[(size_t)(m0 + v + 8 * half) * DMODEL + nt * 16 + l16] = c[nt][v];
  }
}

// out[dst] += Y[src] * invdeg[dst]   (one wave per edge, float4 per lane, f32 atomics -> L2)
__global__ __launch_bounds__(256)
void k_scatter(const int* __restrict__ src, const int* __restrict__ dst,
               const float* __restrict__ Y, const float* __restrict__ invdeg,
               float* __restrict__ out, int E) {
  long gid = (long)blockIdx.x * blockDim.x + threadIdx.x;
  long e   = gid >> 5;
  int lane = (int)(gid & 31);
  if (e >= E) return;
  int s = src[e], d = dst[e];
  float w = invdeg[d];
  float4 y = *((const float4*)(Y + (size_t)s * DMODEL) + lane);
  float* o = out + (size_t)d * DMODEL + lane * 4;
  atomicAdd(o + 0, y.x * w);
  atomicAdd(o + 1, y.y * w);
  atomicAdd(o + 2, y.z * w);
  atomicAdd(o + 3, y.w * w);
}

extern "C" void kernel_launch(void* const* d_in, const int* in_sizes, int n_in,
                              void* d_out, int out_size, void* d_ws, size_t ws_size,
                              hipStream_t stream) {
  const float* x    = (const float*)d_in[0];
  const float* W    = (const float*)d_in[1];
  const float* Wl   = (const float*)d_in[2];
  const float* bias = (const float*)d_in[3];
  const int*   src  = (const int*)d_in[4];
  const int*   dst  = (const int*)d_in[5];

  const int N  = in_sizes[0] / DMODEL;
  const int R  = in_sizes[1] / (DMODEL * DMODEL);
  const int RE = in_sizes[4];
  const int E  = RE / R;

  auto alignUp = [](size_t v) { return (v + 255) & ~(size_t)255; };
  char* wsp = (char*)d_ws;
  unsigned short* Xb  = (unsigned short*)wsp; wsp += alignUp((size_t)N * DMODEL * 2);
  unsigned short* Wbt = (unsigned short*)wsp; wsp += alignUp((size_t)(R + 1) * DMODEL * DMODEL * 2);
  float* Y   = (float*)wsp;                   wsp += alignUp((size_t)N * DMODEL * 4);
  float* deg = (float*)wsp;                   wsp += alignUp((size_t)R * N * 4);
  float* out = (float*)d_out;

  const int B = 256;
  long nOut = (long)N * DMODEL;
  long nDeg = (long)R * N;

  k_zero_f32<<<(int)((nOut + B - 1) / B), B, 0, stream>>>(out, nOut);
  k_zero_f32<<<(int)((nDeg + B - 1) / B), B, 0, stream>>>(deg, nDeg);
  k_cvt_bf16<<<(int)((nOut + B - 1) / B), B, 0, stream>>>(x, Xb, nOut);
  int wtot = (R + 1) * DMODEL * DMODEL;
  k_cvt_w_t<<<(wtot + B - 1) / B, B, 0, stream>>>(W, Wl, Wbt, R);
  k_degree<<<(RE + B - 1) / B, B, 0, stream>>>(dst, deg, RE, E, N);
  k_invdeg<<<(int)((nDeg + B - 1) / B), B, 0, stream>>>(deg, nDeg);

  int waves      = (N + 15) / 16;
  int gemmBlocks = (waves + 7) / 8;            // 8 waves (16-row slabs) per block
  long scThreads = (long)E * 32;
  int scBlocks   = (int)((scThreads + B - 1) / B);

  for (int r = 0; r < R; r++) {
    k_gemm_bf16<false><<<gemmBlocks, B, 0, stream>>>(
        Xb, Wbt + (size_t)r * DMODEL * DMODEL, Y, nullptr, N);
    k_scatter<<<scBlocks, B, 0, stream>>>(src + (size_t)r * E, dst + (size_t)r * E,
                                          Y, deg + (size_t)r * N, out, E);
  }
  k_gemm_bf16<true><<<gemmBlocks, B, 0, stream>>>(
      Xb, Wbt + (size_t)R * DMODEL * DMODEL, out, bias, N);
}
